// LSGNN_79164837200644
// MI455X (gfx1250) — matrix operand order repaired and checked by reading, not compiled, not run on hardware
//
#include <hip/hip_runtime.h>
#include <math.h>

typedef __attribute__((ext_vector_type(16))) __bf16 v16bf;
typedef __attribute__((ext_vector_type(8)))  float  v8f;

#define NN     20000
#define EE     320000
#define ETOT   (EE + NN)
#define INDIM  256
#define HD     128
#define CCLS   40
#define LLAY   3
#define LN_EPS 1e-5f

static __device__ __forceinline__ __bf16 to_bf(float x) { return (__bf16)x; }

// ---------------------------------------------------------------------------
// Generic node-level WMMA GEMM:
//   out = post( [A1 | A2](nrows x K) @ W(K x M) + bias (+res) )
//   post = optional LayerNorm (ln_g/ln_b non-null) then optional ReLU.
// Block: 128 threads (4 waves). Each block computes a 16-row strip x Mpad cols.
// Wave w owns column tiles {w, w+4} (static accumulators, no indexed VGPRs).
// ---------------------------------------------------------------------------
__global__ __launch_bounds__(128) void gemm_node(
    const float* __restrict__ A1, const float* __restrict__ A2, int K1,
    const float* __restrict__ W, int K, int M,
    const float* __restrict__ bias, const float* __restrict__ res,
    const float* __restrict__ ln_g, const float* __restrict__ ln_b,
    float* __restrict__ out, int nrows, int Mpad, int relu_flag)
{
  __shared__ __align__(16) __bf16 aSt[16][32];     // A strip (bf16)
  __shared__ __align__(16) __bf16 wT[128][32];     // W tile, transposed: wT[col][k]
  __shared__ __align__(16) float  cSt[16][128];    // output strip

  const int tid  = threadIdx.x;
  const int lane = tid & 31;
  const int wave = tid >> 5;
  const int hs   = lane >> 4;        // half-wave select
  const int ml   = lane & 15;
  const int row0 = blockIdx.x * 16;
  const int ntile = Mpad >> 4;

  const int t0 = wave;               // first column tile of this wave
  const int t1 = wave + 4;           // second column tile of this wave
  const bool has0 = (t0 < ntile);    // wave-uniform
  const bool has1 = (t1 < ntile);    // wave-uniform

  v8f acc0, acc1;
#pragma unroll
  for (int i = 0; i < 8; ++i) { acc0[i] = 0.f; acc1[i] = 0.f; }

  const int mq = Mpad >> 2;          // float4 chunks per W row

  for (int k0 = 0; k0 < K; k0 += 32) {
    __syncthreads();
    // stage A strip (16 x 32) as one float4 per thread, packed to bf16
    {
      int r  = tid >> 3;             // 0..15
      int kk = (tid & 7) << 2;       // 0,4,...,28
      int grow = row0 + r, k = k0 + kk;
      float4 v = make_float4(0.f, 0.f, 0.f, 0.f);
      if (grow < nrows) {
        const float* src = (k < K1) ? (A1 + (size_t)grow * K1 + k)
                                    : (A2 + (size_t)grow * (K - K1) + (k - K1));
        v = *(const float4*)src;
      }
      aSt[r][kk + 0] = to_bf(v.x);
      aSt[r][kk + 1] = to_bf(v.y);
      aSt[r][kk + 2] = to_bf(v.z);
      aSt[r][kk + 3] = to_bf(v.w);
    }
    // stage W tile transposed: wT[m][kk] = W[(k0+kk)*M + m], float4 loads along m
    for (int q = tid; q < mq * 32; q += 128) {
      int m  = (q % mq) << 2;
      int kk = q / mq;
      int k  = k0 + kk;
      float4 v = make_float4(0.f, 0.f, 0.f, 0.f);
      if (m + 3 < M) {
        v = *(const float4*)(W + (size_t)k * M + m);
      } else {
        float t[4];
#pragma unroll
        for (int j = 0; j < 4; ++j) t[j] = (m + j < M) ? W[(size_t)k * M + m + j] : 0.f;
        v = make_float4(t[0], t[1], t[2], t[3]);
      }
      wT[m + 0][kk] = to_bf(v.x);
      wT[m + 1][kk] = to_bf(v.y);
      wT[m + 2][kk] = to_bf(v.z);
      wT[m + 3][kk] = to_bf(v.w);
    }
    __syncthreads();

    // A fragment (16-bit 16x32): lane row = lane%16; lanes<16 hold K {0..15},
    // lanes>=16 hold K {16..31}  -> one contiguous 16-element run (matches B)
    v16bf af;
#pragma unroll
    for (int e = 0; e < 8; ++e)  af[e] = aSt[ml][e + 16 * hs];
#pragma unroll
    for (int e = 8; e < 16; ++e) af[e] = aSt[ml][e + 16 * hs];

    if (has0) {
      int col = t0 * 16 + ml;
      v16bf bf;
#pragma unroll
      for (int e = 0; e < 16; ++e) bf[e] = wT[col][e + 16 * hs];
      acc0 = __builtin_amdgcn_wmma_f32_16x16x32_bf16(
          false, af, false, bf, (short)0, acc0, false, false);
    }
    if (has1) {
      int col = t1 * 16 + ml;
      v16bf bf;
#pragma unroll
      for (int e = 0; e < 16; ++e) bf[e] = wT[col][e + 16 * hs];
      acc1 = __builtin_amdgcn_wmma_f32_16x16x32_bf16(
          false, af, false, bf, (short)0, acc1, false, false);
    }
  }
  __syncthreads();
  // spill accumulators: D layout — VGPR r: rows r (lanes<16) / r+8 (lanes>=16)
  if (has0) {
    int col = t0 * 16 + ml;
#pragma unroll
    for (int r = 0; r < 8; ++r) cSt[r + 8 * hs][col] = acc0[r];
  }
  if (has1) {
    int col = t1 * 16 + ml;
#pragma unroll
    for (int r = 0; r < 8; ++r) cSt[r + 8 * hs][col] = acc1[r];
  }
  __syncthreads();

  if (ln_g) {  // LayerNorm path (M == 128): one thread per row, 3 passes
    if (tid < 16) {
      int grow = row0 + tid;
      if (grow < nrows) {
        float mu = 0.f;
        for (int c = 0; c < M; ++c) {
          float v = cSt[tid][c];
          if (bias) v += bias[c];
          if (res)  v += res[(size_t)grow * M + c];
          mu += v;
        }
        mu /= (float)M;
        float var = 0.f;
        for (int c = 0; c < M; ++c) {
          float v = cSt[tid][c];
          if (bias) v += bias[c];
          if (res)  v += res[(size_t)grow * M + c];
          float d = v - mu;
          var += d * d;
        }
        var /= (float)M;
        float rs = rsqrtf(var + LN_EPS);
        for (int c = 0; c < M; ++c) {
          float v = cSt[tid][c];
          if (bias) v += bias[c];
          if (res)  v += res[(size_t)grow * M + c];
          float o = (v - mu) * rs * ln_g[c] + ln_b[c];
          if (relu_flag) o = fmaxf(o, 0.f);
          out[(size_t)grow * M + c] = o;
        }
      }
    }
  } else {     // elementwise path
    for (int idx = tid; idx < 16 * Mpad; idx += 128) {
      int r = idx / Mpad, c = idx % Mpad;
      int grow = row0 + r;
      if (grow < nrows && c < M) {
        float v = cSt[r][c];
        if (bias) v += bias[c];
        if (res)  v += res[(size_t)grow * M + c];
        if (relu_flag) v = fmaxf(v, 0.f);
        out[(size_t)grow * M + c] = v;
      }
    }
  }
}

// ---------------------------------------------------------------------------
// Per-layer prep: W1c (128x128 f32, k-major) -> bf16 transposed [col][k] in global
// so edge blocks can raw-copy it into LDS with async loads.
// ---------------------------------------------------------------------------
__global__ __launch_bounds__(256) void w1c_prep(
    const float* __restrict__ W1c, __bf16* __restrict__ wTg)
{
  int idx = blockIdx.x * 256 + threadIdx.x;   // 16384 elements
  if (idx < 128 * 128) {
    int k = idx >> 7, m = idx & 127;
    wTg[m * 128 + k] = to_bf(W1c[idx]);
  }
}

// ---------------------------------------------------------------------------
// Edge kernel: per wave, 16 edges.
//   hmid = relu( |h[dst]-h[src]| @ W1c + tmpA[dst] + tmpB[src] )   (b1 folded in tmpA)
//   score = sigmoid( hmid . w2 + b2 )
//   agg[dst] += score * msgm[src]      (f32 atomics)
// W1c tile arrives via global_load_async_to_lds_b128 (ASYNCcnt-tracked DMA).
// 4 K-steps x 8 column tiles, fully unrolled -> 32 straight-line WMMAs.
// ---------------------------------------------------------------------------
__global__ __launch_bounds__(128) void edge_kernel(
    const float* __restrict__ h,  const float* __restrict__ tA,
    const float* __restrict__ tB, const float* __restrict__ mm,
    const int*   __restrict__ ei, const __bf16* __restrict__ wTg,
    const float* __restrict__ w2, const float* __restrict__ b2p,
    float* __restrict__ agg)
{
  __shared__ __align__(16) __bf16 wT[128][128];     // W1c^T (bf16, 32 KB)
  __shared__ __align__(16) __bf16 dab[4][16][128];  // |h_i - h_j| per wave (16 KB)
  __shared__ int    sS[4][16], sD[4][16];
  __shared__ float  sSc[4][16];

  const int tid  = threadIdx.x;
  const int lane = tid & 31;
  const int w    = tid >> 5;
  const int hs   = lane >> 4;
  const int ml   = lane & 15;
  const int ebase = blockIdx.x * 64 + w * 16;

  // async DMA: 32KB raw copy wTg -> wT (16B chunks, 16 per thread)
  {
    unsigned ldsbase = (unsigned)(uintptr_t)(&wT[0][0]);
#pragma unroll
    for (int it = 0; it < 16; ++it) {
      unsigned chunk = (unsigned)(tid + it * 128) * 16u;
      unsigned loff  = ldsbase + chunk;
      asm volatile("global_load_async_to_lds_b128 %0, %1, %2 offset:0"
                   :: "v"(loff), "v"(chunk), "s"(wTg) : "memory");
    }
  }
  // edge indices (with self loops appended)
  if (lane < 16) {
    int eg = ebase + lane;
    int s = 0, d = 0;
    if (eg < ETOT) {
      if (eg < EE) { s = ei[eg]; d = ei[EE + eg]; }
      else         { s = eg - EE; d = s; }
    }
    sS[w][lane] = s; sD[w][lane] = d;
  }
  __syncthreads();
  // |h_i - h_j| into LDS (bf16): one float4 per lane per row (128 = 32 lanes x 4)
  {
    int c = lane << 2;
#pragma unroll 4
    for (int e = 0; e < 16; ++e) {
      int i = sD[w][e], j = sS[w][e];
      float4 a = *(const float4*)(h + (size_t)i * HD + c);
      float4 b = *(const float4*)(h + (size_t)j * HD + c);
      dab[w][e][c + 0] = to_bf(fabsf(a.x - b.x));
      dab[w][e][c + 1] = to_bf(fabsf(a.y - b.y));
      dab[w][e][c + 2] = to_bf(fabsf(a.z - b.z));
      dab[w][e][c + 3] = to_bf(fabsf(a.w - b.w));
    }
  }
  // wait for the async W tile, then block-wide sync
  asm volatile("s_wait_asynccnt 0x0" ::: "memory");
  __syncthreads();

  v8f acc[8];
#pragma unroll
  for (int t = 0; t < 8; ++t)
#pragma unroll
    for (int r = 0; r < 8; ++r) acc[t][r] = 0.f;

#pragma unroll
  for (int ks = 0; ks < 4; ++ks) {
    const int k0 = ks * 32;
    v16bf af;
#pragma unroll
    for (int e = 0; e < 8; ++e)  af[e] = dab[w][ml][k0 + e + 16 * hs];
#pragma unroll
    for (int e = 8; e < 16; ++e) af[e] = dab[w][ml][k0 + e + 16 * hs];
#pragma unroll
    for (int t = 0; t < 8; ++t) {
      int col = t * 16 + ml;
      v16bf bf;
#pragma unroll
      for (int e = 0; e < 16; ++e) bf[e] = wT[col][k0 + e + 16 * hs];
      acc[t] = __builtin_amdgcn_wmma_f32_16x16x32_bf16(
          false, af, false, bf, (short)0, acc[t], false, false);
    }
  }

  // fused epilogue: bias (tmpA+tmpB), relu, dot with w2 -> per-row partial
  float part[8];
#pragma unroll
  for (int r = 0; r < 8; ++r) part[r] = 0.f;
#pragma unroll
  for (int t = 0; t < 8; ++t) {
    int col = t * 16 + ml;
    float w2c = w2[col];
#pragma unroll
    for (int r = 0; r < 8; ++r) {
      int rw = 8 * hs + r;
      int i = sD[w][rw], j = sS[w][rw];
      float v = acc[t][r] + tA[(size_t)i * HD + col] + tB[(size_t)j * HD + col];
      v = fmaxf(v, 0.f);
      part[r] += v * w2c;
    }
  }
  float b2 = b2p[0];
#pragma unroll
  for (int r = 0; r < 8; ++r) {
    float s = part[r];
#pragma unroll
    for (int off = 1; off < 16; off <<= 1) s += __shfl_xor(s, off, 32);
    if (ml == 0) {
      int rw = 8 * hs + r;
      int eg = ebase + rw;
      sSc[w][rw] = (eg < ETOT) ? (1.f / (1.f + expf(-(s + b2)))) : 0.f;
    }
  }
  __syncthreads();

  // scatter-add: 64 edges x 128 cols, one edge per pass, coalesced
  for (int idx = tid; idx < 64 * HD; idx += 128) {
    int e = idx >> 7, c = idx & 127;
    int we = e >> 4, el = e & 15;
    float sc = sSc[we][el];
    int i = sD[we][el], j = sS[we][el];
    atomicAdd(&agg[(size_t)i * HD + c], sc * mm[(size_t)j * HD + c]);
  }
}

// ---------------------------------------------------------------------------
extern "C" void kernel_launch(void* const* d_in, const int* in_sizes, int n_in,
                              void* d_out, int out_size, void* d_ws, size_t ws_size,
                              hipStream_t stream) {
  (void)in_sizes; (void)n_in; (void)out_size; (void)ws_size;

  const float* x      = (const float*)d_in[0];
  const int*   ei     = (const int*)  d_in[1];
  const float* proj_W = (const float*)d_in[2];
  const float* proj_b = (const float*)d_in[3];
  const float* ln0_g  = (const float*)d_in[4];
  const float* ln0_b  = (const float*)d_in[5];
  const float* sim_W1 = (const float*)d_in[6];   // (L,384,128)
  const float* sim_b1 = (const float*)d_in[7];   // (L,128)
  const float* sim_W2 = (const float*)d_in[8];   // (L,128,1)
  const float* sim_b2 = (const float*)d_in[9];   // (L,1)
  const float* W_msg  = (const float*)d_in[10];  // (L,128,128)
  const float* W_upd  = (const float*)d_in[11];  // (L,256,128)
  const float* b_upd  = (const float*)d_in[12];  // (L,128)
  const float* ln_g   = (const float*)d_in[13];  // (L,128)
  const float* ln_b   = (const float*)d_in[14];  // (L,128)
  const float* cls_W1 = (const float*)d_in[15];  // (128,64)
  const float* cls_b1 = (const float*)d_in[16];
  const float* cls_W2 = (const float*)d_in[17];  // (64,40)
  const float* cls_b2 = (const float*)d_in[18];
  float* out = (float*)d_out;

  const size_t NH = (size_t)NN * HD;
  float* ws  = (float*)d_ws;
  float* hA  = ws;
  float* hB  = ws + NH;
  float* tA  = ws + 2 * NH;
  float* tB  = ws + 3 * NH;
  float* mm  = ws + 4 * NH;
  float* agg = ws + 5 * NH;
  float* z   = ws + 6 * NH;                 // N x 64
  __bf16* wTg = (__bf16*)(z + (size_t)NN * 64);  // 128x128 bf16 (32 KB)

  const int gN = (NN + 15) / 16;          // 1250 row strips
  const int gE = (ETOT + 63) / 64;        // edge blocks
  dim3 blk(128);

  // h = relu(LN(x @ proj_W + proj_b))
  gemm_node<<<gN, blk, 0, stream>>>(x, nullptr, INDIM, proj_W, INDIM, HD,
                                    proj_b, nullptr, ln0_g, ln0_b,
                                    hA, NN, HD, 1);

  float* h = hA; float* hn = hB;
  for (int l = 0; l < LLAY; ++l) {
    const float* W1a = sim_W1 + (size_t)(l * 384 + 0)   * HD;
    const float* W1b = sim_W1 + (size_t)(l * 384 + 128) * HD;
    const float* W1c = sim_W1 + (size_t)(l * 384 + 256) * HD;

    // tmpA = h @ W1a + b1 ; tmpB = h @ W1b ; msgm = h @ W_msg
    gemm_node<<<gN, blk, 0, stream>>>(h, nullptr, HD, W1a, HD, HD,
                                      sim_b1 + l * HD, nullptr, nullptr, nullptr,
                                      tA, NN, HD, 0);
    gemm_node<<<gN, blk, 0, stream>>>(h, nullptr, HD, W1b, HD, HD,
                                      nullptr, nullptr, nullptr, nullptr,
                                      tB, NN, HD, 0);
    gemm_node<<<gN, blk, 0, stream>>>(h, nullptr, HD, W_msg + (size_t)l * HD * HD, HD, HD,
                                      nullptr, nullptr, nullptr, nullptr,
                                      mm, NN, HD, 0);

    hipMemsetAsync(agg, 0, NH * sizeof(float), stream);
    w1c_prep<<<64, 256, 0, stream>>>(W1c, wTg);

    edge_kernel<<<gE, blk, 0, stream>>>(h, tA, tB, mm, ei, wTg,
                                        sim_W2 + l * HD, sim_b2 + l, agg);

    // h' = relu(LN([h|agg] @ W_upd + b_upd + h))
    gemm_node<<<gN, blk, 0, stream>>>(h, agg, HD, W_upd + (size_t)l * 2 * HD * HD, 2 * HD, HD,
                                      b_upd + l * HD, h, ln_g + l * HD, ln_b + l * HD,
                                      hn, NN, HD, 1);
    float* t = h; h = hn; hn = t;
  }

  // logits = relu(h @ cls_W1 + b1) @ cls_W2 + b2
  gemm_node<<<gN, blk, 0, stream>>>(h, nullptr, HD, cls_W1, HD, 64,
                                    cls_b1, nullptr, nullptr, nullptr,
                                    z, NN, 64, 1);
  gemm_node<<<gN, blk, 0, stream>>>(z, nullptr, 64, cls_W2, 64, CCLS,
                                    cls_b2, nullptr, nullptr, nullptr,
                                    out, NN, 48, 0);
}
